// SequenceLabel_41669772705896
// MI455X (gfx1250) — compile-verified, matrix operand order checked by reading, not run on hardware
//
#include <hip/hip_runtime.h>
#include <hip/hip_bf16.h>

#define Bdim 64
#define Sdim 512
#define Hdim 1024
#define Tdim 48

typedef __attribute__((ext_vector_type(16))) _Float16 v16h;
typedef __attribute__((ext_vector_type(8)))  _Float16 v8h;
typedef __attribute__((ext_vector_type(8)))  float    v8f;
typedef __attribute__((ext_vector_type(4)))  float    v4f;

// ---------------------------------------------------------------------------
// Kernel 0: transpose + convert W [H,T] f32 -> Wt [T,H] f16 (192 KB -> 96 KB)
// ---------------------------------------------------------------------------
__global__ void pack_wt_kernel(const float* __restrict__ W, _Float16* __restrict__ Wt) {
    int idx = blockIdx.x * blockDim.x + threadIdx.x;
    if (idx < Tdim * Hdim) {
        int t = idx / Hdim;
        int k = idx - t * Hdim;
        Wt[idx] = (_Float16)W[k * Tdim + t];
    }
}

// ---------------------------------------------------------------------------
// Kernel 1: emissions = seq[B*S, H] x W[H, T] + b  via v_wmma_f32_16x16x32_f16
// One wave per 16(M) x 48(N) tile; K-loop of 32; f32 loads converted to f16.
// ---------------------------------------------------------------------------
__global__ __launch_bounds__(32)
void emis_gemm_kernel(const float* __restrict__ seq, const _Float16* __restrict__ Wt,
                      const float* __restrict__ bias, float* __restrict__ logits) {
    const int lane = threadIdx.x;
    const int lo = lane & 15;
    const int hi = lane >> 4;
    const long m0 = (long)blockIdx.x * 16;
    const float* arow = seq + (m0 + lo) * Hdim;

    v8f acc[3];
    acc[0] = {}; acc[1] = {}; acc[2] = {};

    for (int k0 = 0; k0 < Hdim; k0 += 32) {
        // ---- A fragment: lane (hi,lo) holds M=lo, K = k0+8hi..+7 and k0+16+8hi..+7
        const float* pA = arow + k0 + 8 * hi;
        __builtin_prefetch(pA + 256, 0, 1);     // A stream 8 K-iterations ahead
        v4f a0 = *(const v4f*)(pA);
        v4f a1 = *(const v4f*)(pA + 4);
        v4f a2 = *(const v4f*)(pA + 16);
        v4f a3 = *(const v4f*)(pA + 20);
        v16h a;
        a[0]  = (_Float16)a0[0]; a[1]  = (_Float16)a0[1];
        a[2]  = (_Float16)a0[2]; a[3]  = (_Float16)a0[3];
        a[4]  = (_Float16)a1[0]; a[5]  = (_Float16)a1[1];
        a[6]  = (_Float16)a1[2]; a[7]  = (_Float16)a1[3];
        a[8]  = (_Float16)a2[0]; a[9]  = (_Float16)a2[1];
        a[10] = (_Float16)a2[2]; a[11] = (_Float16)a2[3];
        a[12] = (_Float16)a3[0]; a[13] = (_Float16)a3[1];
        a[14] = (_Float16)a3[2]; a[15] = (_Float16)a3[3];

        #pragma unroll
        for (int n = 0; n < 3; ++n) {
            // ---- B fragment: lane holds N = 16n+lo, same K runs; Wt is [T][H] f16
            const _Float16* pB = Wt + (long)(n * 16 + lo) * Hdim + k0 + 8 * hi;
            v8h b0 = *(const v8h*)(pB);
            v8h b1 = *(const v8h*)(pB + 16);
            v16h bb = __builtin_shufflevector(b0, b1, 0, 1, 2, 3, 4, 5, 6, 7,
                                                      8, 9, 10, 11, 12, 13, 14, 15);
            acc[n] = __builtin_amdgcn_wmma_f32_16x16x32_f16(
                false, a, false, bb, (short)0, acc[n], false, false);
        }
    }

    // ---- epilogue: D VGPR r <-> row 8*hi + r, col 16n + lo
    #pragma unroll
    for (int n = 0; n < 3; ++n) {
        const int col = n * 16 + lo;
        const float bv = bias[col];
        const long rbase = m0 + 8 * hi;
        #pragma unroll
        for (int r = 0; r < 8; ++r)
            logits[(rbase + r) * Tdim + col] = acc[n][r] + bv;
    }
}

// ---------------------------------------------------------------------------
// Kernel 2: CRF forward (denominator) + gold-path numerator, one block/batch.
// Exp-space recursion: alpha kept normalized via running offset; exp(trans)
// column resident in 48 VGPRs per thread (fully unrolled FMA matvec).
// ---------------------------------------------------------------------------
__global__ __launch_bounds__(64)
void crf_fwd_kernel(const float* __restrict__ logits, const int* __restrict__ tags,
                    const unsigned char* __restrict__ mask,
                    const float* __restrict__ startt, const float* __restrict__ endt,
                    const float* __restrict__ trans, float* __restrict__ batchLoss) {
    __shared__ float pbuf[Tdim];
    __shared__ float an[Tdim];
    __shared__ float red[64];
    __shared__ int   redi[64];

    const int b = blockIdx.x, tid = threadIdx.x;
    const float* em = logits + (size_t)b * Sdim * Tdim;
    const int* tg = tags + b * Sdim;
    const unsigned char* mk = mask + b * Sdim;

    // L = popcount(mask row) -- mask is a contiguous prefix by construction
    int cnt = 0;
    for (int s = tid; s < Sdim; s += 64) cnt += mk[s] ? 1 : 0;
    redi[tid] = cnt;
    __syncthreads();
    for (int w = 32; w > 0; w >>= 1) { if (tid < w) redi[tid] += redi[tid + w]; __syncthreads(); }
    const int L = redi[0];
    __syncthreads();

    // per-thread register column of exp(trans[:, tid])
    float etr[Tdim];
    if (tid < Tdim) {
        #pragma unroll
        for (int i = 0; i < Tdim; ++i) etr[i] = __expf(trans[i * Tdim + tid]);
    }

    float off = 0.0f;
    if (tid < Tdim) an[tid] = startt[tid] + em[tid];
    __syncthreads();

    for (int s = 1; s < L; ++s) {
        const float kap = an[0];                 // cheap normalizer; spread bounded
        if (tid < Tdim) pbuf[tid] = __expf(an[tid] - kap);
        off += kap;
        __syncthreads();
        float aout = 0.0f;
        if (tid < Tdim) {
            float s0 = 0.f, s1 = 0.f, s2 = 0.f, s3 = 0.f;
            #pragma unroll
            for (int i = 0; i < Tdim; i += 4) {
                s0 = fmaf(pbuf[i + 0], etr[i + 0], s0);
                s1 = fmaf(pbuf[i + 1], etr[i + 1], s1);
                s2 = fmaf(pbuf[i + 2], etr[i + 2], s2);
                s3 = fmaf(pbuf[i + 3], etr[i + 3], s3);
            }
            aout = __logf((s0 + s1) + (s2 + s3)) + em[(size_t)s * Tdim + tid];
        }
        __syncthreads();
        if (tid < Tdim) an[tid] = aout;
        __syncthreads();
    }

    // den = off + logsumexp(an + end)
    float v = (tid < Tdim) ? (an[tid] + endt[tid]) : -3.0e38f;
    red[tid] = v;
    __syncthreads();
    for (int w = 32; w > 0; w >>= 1) { if (tid < w) red[tid] = fmaxf(red[tid], red[tid + w]); __syncthreads(); }
    const float mx = red[0];
    __syncthreads();
    red[tid] = (tid < Tdim) ? __expf(v - mx) : 0.0f;
    __syncthreads();
    for (int w = 32; w > 0; w >>= 1) { if (tid < w) red[tid] += red[tid + w]; __syncthreads(); }
    const float den = off + mx + __logf(red[0]);
    __syncthreads();

    // numerator: parallel gather over gold path
    float part = 0.0f;
    for (int s = tid + 1; s < Sdim; s += 64)
        if (mk[s]) part += trans[tg[s - 1] * Tdim + tg[s]] + em[(size_t)s * Tdim + tg[s]];
    red[tid] = part;
    __syncthreads();
    for (int w = 32; w > 0; w >>= 1) { if (tid < w) red[tid] += red[tid + w]; __syncthreads(); }
    if (tid == 0) {
        float num = startt[tg[0]] + em[tg[0]] + red[0] + endt[tg[L - 1]];
        batchLoss[b] = num - den;
    }
}

// ---------------------------------------------------------------------------
// Kernel 3: Viterbi decode, one block/batch; u8 backpointers in workspace,
// first-max tie-break matching jnp.argmax; serial backtrace by thread 0.
// ---------------------------------------------------------------------------
__global__ __launch_bounds__(64)
void crf_viterbi_kernel(const float* __restrict__ logits, const unsigned char* __restrict__ mask,
                        const float* __restrict__ startt, const float* __restrict__ endt,
                        const float* __restrict__ trans, unsigned char* __restrict__ hist,
                        float* __restrict__ out) {
    __shared__ float sc[Tdim];
    __shared__ int   redi[64];
    __shared__ unsigned char path[Sdim];

    const int b = blockIdx.x, tid = threadIdx.x;
    const float* em = logits + (size_t)b * Sdim * Tdim;
    const unsigned char* mk = mask + b * Sdim;

    int cnt = 0;
    for (int s = tid; s < Sdim; s += 64) cnt += mk[s] ? 1 : 0;
    redi[tid] = cnt;
    __syncthreads();
    for (int w = 32; w > 0; w >>= 1) { if (tid < w) redi[tid] += redi[tid + w]; __syncthreads(); }
    const int L = redi[0];
    __syncthreads();

    float trc[Tdim];
    if (tid < Tdim) {
        #pragma unroll
        for (int i = 0; i < Tdim; ++i) trc[i] = trans[i * Tdim + tid];
        sc[tid] = startt[tid] + em[tid];
    }
    __syncthreads();

    unsigned char* hb = hist + (size_t)b * (Sdim - 1) * Tdim;

    for (int s = 1; s < L; ++s) {
        float nsc = 0.0f;
        if (tid < Tdim) {
            float best = sc[0] + trc[0];
            int bi = 0;
            #pragma unroll
            for (int i = 1; i < Tdim; ++i) {
                float c = sc[i] + trc[i];
                if (c > best) { best = c; bi = i; }   // strict > keeps first max
            }
            nsc = best + em[(size_t)s * Tdim + tid];
            hb[(size_t)(s - 1) * Tdim + tid] = (unsigned char)bi;
        }
        __syncthreads();
        if (tid < Tdim) sc[tid] = nsc;
        __syncthreads();
    }

    if (tid == 0) {
        float best = sc[0] + endt[0];
        int cur = 0;
        for (int i = 1; i < Tdim; ++i) {
            float c = sc[i] + endt[i];
            if (c > best) { best = c; cur = i; }
        }
        path[L - 1] = (unsigned char)cur;
        for (int s = L - 2; s >= 0; --s) {
            cur = hb[(size_t)s * Tdim + cur];
            path[s] = (unsigned char)cur;
        }
    }
    __syncthreads();

    float* ob = out + (size_t)b * Sdim;
    for (int s = tid; s < Sdim; s += 64)
        ob[s] = (s < L) ? (float)path[s] : 0.0f;     // reference zeroes padding
}

// ---------------------------------------------------------------------------
// Kernel 4: loss = -sum_b (num_b - den_b)  ->  d_out[B*S]
// ---------------------------------------------------------------------------
__global__ void loss_reduce_kernel(const float* __restrict__ batchLoss, float* __restrict__ out) {
    __shared__ float red[64];
    const int tid = threadIdx.x;
    red[tid] = batchLoss[tid];
    __syncthreads();
    for (int w = 32; w > 0; w >>= 1) { if (tid < w) red[tid] += red[tid + w]; __syncthreads(); }
    if (tid == 0) out[Bdim * Sdim] = -red[0];
}

// ---------------------------------------------------------------------------
extern "C" void kernel_launch(void* const* d_in, const int* in_sizes, int n_in,
                              void* d_out, int out_size, void* d_ws, size_t ws_size,
                              hipStream_t stream) {
    const float* seq   = (const float*)d_in[0];          // [B,S,H] f32
    const float* W     = (const float*)d_in[1];          // [H,T]   f32
    const float* bias  = (const float*)d_in[2];          // [T]     f32
    const float* stt   = (const float*)d_in[3];          // [T]     f32
    const float* ent   = (const float*)d_in[4];          // [T]     f32
    const float* trans = (const float*)d_in[5];          // [T,T]   f32
    const unsigned char* mask = (const unsigned char*)d_in[6];  // [B,S] bool (u8)
    const int* tags    = (const int*)d_in[7];            // [B,S]   i32

    float* out = (float*)d_out;                          // [B*S] pred + [1] loss

    char* ws = (char*)d_ws;
    const size_t logits_bytes = (size_t)Bdim * Sdim * Tdim * sizeof(float);   // 6 291 456
    const size_t wt_bytes     = (size_t)Tdim * Hdim * sizeof(_Float16);       //    98 304
    const size_t bl_bytes     = (size_t)Bdim * sizeof(float);                 //       256
    float*         logits = (float*)ws;
    _Float16*      Wt     = (_Float16*)(ws + logits_bytes);
    float*         bl     = (float*)(ws + logits_bytes + wt_bytes);
    unsigned char* hist   = (unsigned char*)(ws + logits_bytes + wt_bytes + bl_bytes);

    pack_wt_kernel<<<(Tdim * Hdim + 255) / 256, 256, 0, stream>>>(W, Wt);
    emis_gemm_kernel<<<(Bdim * Sdim) / 16, 32, 0, stream>>>(seq, Wt, bias, logits);
    crf_fwd_kernel<<<Bdim, 64, 0, stream>>>(logits, tags, mask, stt, ent, trans, bl);
    crf_viterbi_kernel<<<Bdim, 64, 0, stream>>>(logits, mask, stt, ent, trans, hist, out);
    loss_reduce_kernel<<<1, 64, 0, stream>>>(bl, out);
}
